// MultiHeadSelfAttention_11003706212808
// MI455X (gfx1250) — compile-verified, matrix operand order checked by reading, not run on hardware
//
#include <hip/hip_runtime.h>
#include <hip/hip_bf16.h>

// ---------------------------------------------------------------------------
// MHSA for gfx1250: bf16 WMMA, double-buffered async global->LDS staging of
// all shared/streamed tiles (weights, K, V, mask), flash-style online softmax.
// B=2, L=2048, DIM=1024, H=16, D=64.
// ---------------------------------------------------------------------------

#define N_HEADS 16
#define HEAD_DIM 64
#define DIM 1024
#define BATCH 2
#define SEQ 2048

typedef __attribute__((ext_vector_type(16))) __bf16 v16bf;
typedef __attribute__((ext_vector_type(8)))  float  v8f;
typedef __attribute__((ext_vector_type(4)))  float  v4f;
typedef __attribute__((ext_vector_type(4)))  int    v4i;

#define WMMA_BF16(A, B, C) \
  __builtin_amdgcn_wmma_f32_16x16x32_bf16(false, (A), false, (B), (short)0, (C), false, false)

// --------------------------- async staging path ----------------------------
#if __has_builtin(__builtin_amdgcn_global_load_async_to_lds_b128) && \
    __has_builtin(__builtin_amdgcn_s_wait_asynccnt)
#define ASYNC_STAGE 1
#else
#define ASYNC_STAGE 0
#endif

// Builtin signature (confirmed by hipcc diagnostic): v4i addrspace(1)* src,
// v4i addrspace(3)* dst, imm offset, imm cpol.
typedef __attribute__((address_space(1))) v4i as1_v4i;
typedef __attribute__((address_space(3))) v4i as3_v4i;

// Copy 16 bytes global -> LDS. Async (ASYNCcnt) when available, else a plain
// load + ds_store (made visible by the block barrier).
__device__ __forceinline__ void stage_b128(const void* gsrc, void* ldst) {
#if ASYNC_STAGE
  __builtin_amdgcn_global_load_async_to_lds_b128(
      (as1_v4i*)(unsigned long long)gsrc,
      (as3_v4i*)(unsigned int)(unsigned long long)ldst,
      0, 0);
#else
  *(v4f*)ldst = *(const v4f*)gsrc;
#endif
}

#if ASYNC_STAGE
#define STAGE_WAIT(n) __builtin_amdgcn_s_wait_asynccnt(n)
#else
#define STAGE_WAIT(n) ((void)0)
#endif

// Barrier between double-buffer phases.  With async staging, LDS-tile
// readiness is enforced by s_wait_asynccnt (producer side) and every ds_load
// issued before the end-barrier is consumed by a WMMA before it (dscnt
// drained), so a bare split barrier suffices -- no vmem fence, letting
// next-iteration global A-fragment loads stay in flight across the barrier.
// The fallback (ds_store staging) needs __syncthreads()'s fence.
#if ASYNC_STAGE && __has_builtin(__builtin_amdgcn_s_barrier_signal) && \
    __has_builtin(__builtin_amdgcn_s_barrier_wait)
#define LIGHT_BARRIER 1
#else
#define LIGHT_BARRIER 0
#endif

__device__ __forceinline__ void tile_barrier() {
#if LIGHT_BARRIER
  __builtin_amdgcn_s_barrier_signal(-1);
  __builtin_amdgcn_s_barrier_wait(-1);
#else
  __syncthreads();
#endif
}

// Load a 16x32 WMMA fragment in "NT" striping: lane L supplies row (L&15),
// K-offset base (L>>4)*8; elements 0..7 at +0, 8..15 at +16 (per ISA 16-bit
// A/B VGPR layout). Two b128 loads (global or ds depending on pointer).
__device__ __forceinline__ v16bf load_frag_nt(const __bf16* base, int stride) {
  const int lane = threadIdx.x & 31;
  const __bf16* p = base + (size_t)(lane & 15) * stride + ((lane >> 4) << 3);
  v16bf f;
  ((v4f*)&f)[0] = *(const v4f*)(p);
  ((v4f*)&f)[1] = *(const v4f*)(p + 16);
  return f;
}

// ---------------------------------------------------------------------------
// Stage 0: fp32 -> bf16 conversion (vectorized x4)
// ---------------------------------------------------------------------------
__global__ void cvt_f32_bf16_x4(const float* __restrict__ src,
                                __bf16* __restrict__ dst, int n4) {
  int i = blockIdx.x * blockDim.x + threadIdx.x;
  if (i < n4) {
    v4f v = ((const v4f*)src)[i];
    dst[4 * i + 0] = (__bf16)v[0];
    dst[4 * i + 1] = (__bf16)v[1];
    dst[4 * i + 2] = (__bf16)v[2];
    dst[4 * i + 3] = (__bf16)v[3];
  }
}

// ---------------------------------------------------------------------------
// Stage 1: QKV projection.  out[m,n] = sum_k X[m,k] * W[n,k]  (NT GEMM).
// Block = 128 threads (4 waves). Block tile 128(m) x 64(n); each wave owns
// 32(m) x 64(n) -> 8 wmma per 32-deep k-step. The 64x32 weight tile is
// staged once per block into double-buffered LDS (straight-line, no
// divergent staging loop); A fragments are double-buffered in registers.
// ---------------------------------------------------------------------------
__global__ void __launch_bounds__(128) qkv_gemm(
    const __bf16* __restrict__ Xb,
    const __bf16* __restrict__ Wqb, const __bf16* __restrict__ Wkb,
    const __bf16* __restrict__ Wvb,
    __bf16* __restrict__ Qb, __bf16* __restrict__ Kb,
    __bf16* __restrict__ Vtb) {
  __shared__ __attribute__((aligned(16))) __bf16 Wt[2][64 * 32];  // [n][k]

  const int tid  = threadIdx.x;
  const int wave = tid >> 5, lane = tid & 31;
  const int half = lane >> 4, nl = lane & 15;
  const int m0 = blockIdx.x * 128 + wave * 32;
  const int nb = blockIdx.y * 64;
  const int z  = blockIdx.z;
  const __bf16* W = (z == 0) ? Wqb : (z == 1) ? Wkb : Wvb;

  // 64 rows x 32 halfs (64B) = 256 x 16B segments; 2 per thread, no branches.
  const int sr = tid >> 2, sc = (tid & 3) << 3;
  auto stageB = [&](int buf, int k0) {
    stage_b128(W + (size_t)(nb + sr) * DIM + k0 + sc, &Wt[buf][sr * 32 + sc]);
    stage_b128(W + (size_t)(nb + sr + 32) * DIM + k0 + sc,
               &Wt[buf][(sr + 32) * 32 + sc]);
  };
  auto loadA = [&](int k0, v16bf* a) {
    a[0] = load_frag_nt(Xb + (size_t)m0 * DIM + k0, DIM);
    a[1] = load_frag_nt(Xb + (size_t)(m0 + 16) * DIM + k0, DIM);
  };

  v8f acc[2][4] = {};
  v16bf aC[2], aN[2];
  stageB(0, 0);
  loadA(0, aC);

  const int KSTEPS = DIM / 32;
  for (int i = 0; i < KSTEPS; ++i) {
    const int cur = i & 1;
    if (i + 1 < KSTEPS) {
      stageB(cur ^ 1, (i + 1) * 32);
      loadA((i + 1) * 32, aN);
      STAGE_WAIT(2);                 // current stage done; next in flight
    } else {
      STAGE_WAIT(0);
    }
    tile_barrier();                  // buf[cur] visible block-wide
#pragma unroll
    for (int t = 0; t < 4; ++t) {
      v16bf b = load_frag_nt(&Wt[cur][(t * 16) * 32], 32);
      acc[0][t] = WMMA_BF16(aC[0], b, acc[0][t]);
      acc[1][t] = WMMA_BF16(aC[1], b, acc[1][t]);
    }
    tile_barrier();                  // done reading buf[cur] before overwrite
    aC[0] = aN[0];
    aC[1] = aN[1];
  }

  const int h = nb >> 6;  // 16-wide n-tiles never straddle a head
#pragma unroll
  for (int mc = 0; mc < 2; ++mc) {
#pragma unroll
    for (int t = 0; t < 4; ++t) {
      const int d = (nb & 63) + t * 16 + nl;
#pragma unroll
      for (int e = 0; e < 8; ++e) {
        const int m = m0 + mc * 16 + e + 8 * half;
        const int bb = m >> 11, l = m & (SEQ - 1);
        __bf16 v = (__bf16)acc[mc][t][e];
        if (z == 2)
          Vtb[(((size_t)bb * N_HEADS + h) * HEAD_DIM + d) * SEQ + l] = v;
        else if (z == 0)
          Qb[(((size_t)bb * N_HEADS + h) * SEQ + l) * HEAD_DIM + d] = v;
        else
          Kb[(((size_t)bb * N_HEADS + h) * SEQ + l) * HEAD_DIM + d] = v;
      }
    }
  }
}

// ---------------------------------------------------------------------------
// Stage 2: flash attention.  Block = 4 waves, one (b,h), 64 q-rows (16/wave).
// Per 32-wide k-step: K (32x64 bf16), V^T (64x32 bf16) AND the 64x32 int
// mask tile are staged cooperatively into double-buffered LDS via async
// copies; S = Q@K^T (4 wmma), mask+scale from LDS, online softmax
// (xor-shuffle reductions within each 16-lane half), P through per-wave LDS
// tile (LDS in-order within a wave), then P@V (4 wmma).
// ---------------------------------------------------------------------------
__global__ void __launch_bounds__(128) flash_attn(
    const __bf16* __restrict__ Qb, const __bf16* __restrict__ Kb,
    const __bf16* __restrict__ Vtb, const int* __restrict__ mask,
    __bf16* __restrict__ Cb) {
  __shared__ __attribute__((aligned(16))) __bf16 KT[2][32 * 64];  // [n][d]
  __shared__ __attribute__((aligned(16))) __bf16 VT[2][64 * 32];  // [d][k]
  __shared__ __attribute__((aligned(16))) int    MT[2][64 * 32];  // [q][k]
  __shared__ __attribute__((aligned(16))) __bf16 PT[4][16 * 32];  // per wave

  const int tid  = threadIdx.x;
  const int wave = tid >> 5, lane = tid & 31;
  const int half = lane >> 4, nl = lane & 15;
  const int b = blockIdx.z, h = blockIdx.y;
  const int mblk = blockIdx.x * 64;
  const int m0 = mblk + wave * 16;

  const __bf16* Qh = Qb + (((size_t)b * N_HEADS + h) * SEQ) * HEAD_DIM;
  const __bf16* Kh = Kb + (((size_t)b * N_HEADS + h) * SEQ) * HEAD_DIM;
  const __bf16* Vh = Vtb + (((size_t)b * N_HEADS + h) * HEAD_DIM) * SEQ;
  const int* maskb = mask + (size_t)b * SEQ * SEQ;

  // Straight-line staging, 8 x b128 per thread per stage:
  //   K: 32 rows x 128B  = 256 segs (2/thread)
  //   V: 64 rows x  64B  = 256 segs (2/thread)
  //   M: 64 rows x 128B  = 512 segs (4/thread)
  const int kr = tid >> 3, kc = (tid & 7) << 3;  // halfs
  const int vr = tid >> 2, vc = (tid & 3) << 3;  // halfs
  const int mr = tid >> 3, mc4 = (tid & 7) << 2; // ints
  auto stageKVM = [&](int buf, int k0) {
    stage_b128(Kh + (size_t)(k0 + kr) * HEAD_DIM + kc, &KT[buf][kr * 64 + kc]);
    stage_b128(Kh + (size_t)(k0 + kr + 16) * HEAD_DIM + kc,
               &KT[buf][(kr + 16) * 64 + kc]);
    stage_b128(Vh + (size_t)vr * SEQ + k0 + vc, &VT[buf][vr * 32 + vc]);
    stage_b128(Vh + (size_t)(vr + 32) * SEQ + k0 + vc,
               &VT[buf][(vr + 32) * 32 + vc]);
    stage_b128(maskb + (size_t)(mblk + mr) * SEQ + k0 + mc4,
               &MT[buf][mr * 32 + mc4]);
    stage_b128(maskb + (size_t)(mblk + mr + 16) * SEQ + k0 + mc4,
               &MT[buf][(mr + 16) * 32 + mc4]);
    stage_b128(maskb + (size_t)(mblk + mr + 32) * SEQ + k0 + mc4,
               &MT[buf][(mr + 32) * 32 + mc4]);
    stage_b128(maskb + (size_t)(mblk + mr + 48) * SEQ + k0 + mc4,
               &MT[buf][(mr + 48) * 32 + mc4]);
  };

  // Q tile (16 x 64) = two 32-deep A fragments, loop-invariant.
  v16bf a0 = load_frag_nt(Qh + (size_t)m0 * HEAD_DIM, HEAD_DIM);
  v16bf a1 = load_frag_nt(Qh + (size_t)m0 * HEAD_DIM + 32, HEAD_DIM);

  float mrow[8], lrow[8];
#pragma unroll
  for (int e = 0; e < 8; ++e) { mrow[e] = -1.0e30f; lrow[e] = 0.0f; }
  v8f acc[4] = {};

  stageKVM(0, 0);
  const int STEPS = SEQ / 32;
  for (int i = 0; i < STEPS; ++i) {
    const int cur = i & 1;
    const int k0 = i * 32;
    if (i + 1 < STEPS) {
      stageKVM(cur ^ 1, k0 + 32);
      STAGE_WAIT(8);
    } else {
      STAGE_WAIT(0);
    }
    tile_barrier();

    // S tiles: n = k0..k0+15 and k0+16..k0+31, reduction over d (2 x 32).
    v8f c0 = {}, c1 = {};
    {
      v16bf bk = load_frag_nt(&KT[cur][0], 64);
      c0 = WMMA_BF16(a0, bk, c0);
      bk = load_frag_nt(&KT[cur][32], 64);
      c0 = WMMA_BF16(a1, bk, c0);
      bk = load_frag_nt(&KT[cur][16 * 64], 64);
      c1 = WMMA_BF16(a0, bk, c1);
      bk = load_frag_nt(&KT[cur][16 * 64 + 32], 64);
      c1 = WMMA_BF16(a1, bk, c1);
    }

#pragma unroll
    for (int e = 0; e < 8; ++e) {
      const int rl = wave * 16 + e + 8 * half;  // q-row within block tile
      float s0 = c0[e] * 0.125f;  // 1/sqrt(64)
      float s1 = c1[e] * 0.125f;
      if (MT[cur][rl * 32 + nl]      == 0) s0 = -1.0e30f;
      if (MT[cur][rl * 32 + nl + 16] == 0) s1 = -1.0e30f;

      float t = fmaxf(s0, s1);
      t = fmaxf(t, __shfl_xor(t, 1, 32));
      t = fmaxf(t, __shfl_xor(t, 2, 32));
      t = fmaxf(t, __shfl_xor(t, 4, 32));
      t = fmaxf(t, __shfl_xor(t, 8, 32));

      const float mnew = fmaxf(mrow[e], t);
      const float corr = __expf(mrow[e] - mnew);
      const float p0 = __expf(s0 - mnew);
      const float p1 = __expf(s1 - mnew);
      float rs = p0 + p1;
      rs += __shfl_xor(rs, 1, 32);
      rs += __shfl_xor(rs, 2, 32);
      rs += __shfl_xor(rs, 4, 32);
      rs += __shfl_xor(rs, 8, 32);
      lrow[e] = lrow[e] * corr + rs;
      mrow[e] = mnew;
#pragma unroll
      for (int t4 = 0; t4 < 4; ++t4) acc[t4][e] = acc[t4][e] * corr;

      const int r = e + 8 * half;  // row within 16x32 P tile
      PT[wave][r * 32 + nl]      = (__bf16)p0;
      PT[wave][r * 32 + nl + 16] = (__bf16)p1;
    }

    // LDS is in-order within a wave: reread P as an A fragment.
    v16bf pA = load_frag_nt(&PT[wave][0], 32);
#pragma unroll
    for (int t4 = 0; t4 < 4; ++t4) {
      v16bf bv = load_frag_nt(&VT[cur][(t4 * 16) * 32], 32);
      acc[t4] = WMMA_BF16(pA, bv, acc[t4]);
    }
    tile_barrier();  // all waves done with buf[cur] before it is restaged
  }

  // Epilogue: ctx = acc / l, bf16 into [B*L, H*D] for the output GEMM.
#pragma unroll
  for (int t4 = 0; t4 < 4; ++t4) {
    const int d = t4 * 16 + nl;
#pragma unroll
    for (int e = 0; e < 8; ++e) {
      const int m = m0 + e + 8 * half;
      const float v = acc[t4][e] / lrow[e];
      Cb[((size_t)b * SEQ + m) * DIM + h * HEAD_DIM + d] = (__bf16)v;
    }
  }
}

// ---------------------------------------------------------------------------
// Stage 3: output projection.  out[m,n] = sum_k Ctx[m,k] * Wo[n,k], f32 out.
// Same structure as qkv_gemm.
// ---------------------------------------------------------------------------
__global__ void __launch_bounds__(128) out_gemm(
    const __bf16* __restrict__ Cb, const __bf16* __restrict__ Wob,
    float* __restrict__ out) {
  __shared__ __attribute__((aligned(16))) __bf16 Wt[2][64 * 32];

  const int tid  = threadIdx.x;
  const int wave = tid >> 5, lane = tid & 31;
  const int half = lane >> 4, nl = lane & 15;
  const int m0 = blockIdx.x * 128 + wave * 32;
  const int nb = blockIdx.y * 64;

  const int sr = tid >> 2, sc = (tid & 3) << 3;
  auto stageB = [&](int buf, int k0) {
    stage_b128(Wob + (size_t)(nb + sr) * DIM + k0 + sc, &Wt[buf][sr * 32 + sc]);
    stage_b128(Wob + (size_t)(nb + sr + 32) * DIM + k0 + sc,
               &Wt[buf][(sr + 32) * 32 + sc]);
  };
  auto loadA = [&](int k0, v16bf* a) {
    a[0] = load_frag_nt(Cb + (size_t)m0 * DIM + k0, DIM);
    a[1] = load_frag_nt(Cb + (size_t)(m0 + 16) * DIM + k0, DIM);
  };

  v8f acc[2][4] = {};
  v16bf aC[2], aN[2];
  stageB(0, 0);
  loadA(0, aC);

  const int KSTEPS = DIM / 32;
  for (int i = 0; i < KSTEPS; ++i) {
    const int cur = i & 1;
    if (i + 1 < KSTEPS) {
      stageB(cur ^ 1, (i + 1) * 32);
      loadA((i + 1) * 32, aN);
      STAGE_WAIT(2);
    } else {
      STAGE_WAIT(0);
    }
    tile_barrier();
#pragma unroll
    for (int t = 0; t < 4; ++t) {
      v16bf b = load_frag_nt(&Wt[cur][(t * 16) * 32], 32);
      acc[0][t] = WMMA_BF16(aC[0], b, acc[0][t]);
      acc[1][t] = WMMA_BF16(aC[1], b, acc[1][t]);
    }
    tile_barrier();
    aC[0] = aN[0];
    aC[1] = aN[1];
  }

#pragma unroll
  for (int mc = 0; mc < 2; ++mc) {
#pragma unroll
    for (int t = 0; t < 4; ++t) {
#pragma unroll
      for (int e = 0; e < 8; ++e) {
        const int m = m0 + mc * 16 + e + 8 * half;
        out[(size_t)m * DIM + nb + t * 16 + nl] = acc[mc][t][e];
      }
    }
  }
}

// ---------------------------------------------------------------------------
// Host launch.  d_in order: X, attention_mask, Wq, Wk, Wv, Wo.
// Workspace layout (bytes):
//   Xb 8MB | Wqb 2MB | Wkb 2MB | Wvb 2MB | Wob 2MB |
//   Qb 8MB | Kb 8MB | Vtb 8MB | Cb 8MB   = 48MB total
// ---------------------------------------------------------------------------
extern "C" void kernel_launch(void* const* d_in, const int* in_sizes, int n_in,
                              void* d_out, int out_size, void* d_ws, size_t ws_size,
                              hipStream_t stream) {
  const float* X    = (const float*)d_in[0];
  const int*   mask = (const int*)d_in[1];
  const float* Wq   = (const float*)d_in[2];
  const float* Wk   = (const float*)d_in[3];
  const float* Wv   = (const float*)d_in[4];
  const float* Wo   = (const float*)d_in[5];
  float* out = (float*)d_out;

  char* ws = (char*)d_ws;
  const size_t MB = 1024 * 1024;
  __bf16* Xb  = (__bf16*)(ws + 0);
  __bf16* Wqb = (__bf16*)(ws + 8 * MB);
  __bf16* Wkb = (__bf16*)(ws + 10 * MB);
  __bf16* Wvb = (__bf16*)(ws + 12 * MB);
  __bf16* Wob = (__bf16*)(ws + 14 * MB);
  __bf16* Qb  = (__bf16*)(ws + 16 * MB);
  __bf16* Kb  = (__bf16*)(ws + 24 * MB);
  __bf16* Vtb = (__bf16*)(ws + 32 * MB);
  __bf16* Cb  = (__bf16*)(ws + 40 * MB);

  const int nX = BATCH * SEQ * DIM;  // 4,194,304
  const int nW = DIM * DIM;          // 1,048,576

  cvt_f32_bf16_x4<<<(nX / 4 + 255) / 256, 256, 0, stream>>>(X, Xb, nX / 4);
  cvt_f32_bf16_x4<<<(nW / 4 + 255) / 256, 256, 0, stream>>>(Wq, Wqb, nW / 4);
  cvt_f32_bf16_x4<<<(nW / 4 + 255) / 256, 256, 0, stream>>>(Wk, Wkb, nW / 4);
  cvt_f32_bf16_x4<<<(nW / 4 + 255) / 256, 256, 0, stream>>>(Wv, Wvb, nW / 4);
  cvt_f32_bf16_x4<<<(nW / 4 + 255) / 256, 256, 0, stream>>>(Wo, Wob, nW / 4);

  // QKV: block tile 128m x 64n: grid (4096/128, 1024/64, 3).
  qkv_gemm<<<dim3(32, 16, 3), 128, 0, stream>>>(Xb, Wqb, Wkb, Wvb, Qb, Kb, Vtb);

  // Flash attention: 32 q-blocks of 64 rows x 16 heads x 2 batches.
  flash_attn<<<dim3(SEQ / 64, N_HEADS, BATCH), 128, 0, stream>>>(Qb, Kb, Vtb, mask, Cb);

  // Output projection -> f32 d_out: grid (4096/128, 1024/64).
  out_gemm<<<dim3(32, 16), 128, 0, stream>>>(Cb, Wob, out);
}
